// ExactQCNN9Open_32040456028229
// MI455X (gfx1250) — compile-verified, matrix operand order checked by reading, not run on hardware
//
#include <hip/hip_runtime.h>

typedef __attribute__((ext_vector_type(16))) _Float16 v16h;
typedef __attribute__((ext_vector_type(8)))  _Float16 v8h;
typedef __attribute__((ext_vector_type(4)))  _Float16 v4h;
typedef __attribute__((ext_vector_type(8)))  float    v8f;

#define NQ   9
#define QDIM 512
#define SQ2F 0.70710678118654752440f

__device__ __forceinline__ int wmask(int w) { return 1 << (8 - w); }

// ---------------- gate helpers: one thread per amplitude, state in LDS -------
__device__ __forceinline__ void g_h(float* sr, float* si, int i, int w) {
  const int m = wmask(w), p = i ^ m;
  __syncthreads();
  float ar = sr[i], ai = si[i], br = sr[p], bi = si[p];
  __syncthreads();
  if (i & m) { sr[i] = (br - ar) * SQ2F; si[i] = (bi - ai) * SQ2F; }
  else       { sr[i] = (ar + br) * SQ2F; si[i] = (ai + bi) * SQ2F; }
}

__device__ __forceinline__ void g_ph(float* sr, float* si, int i, int w,
                                     float pr, float pi) {
  if (i & wmask(w)) {
    float r = sr[i], im = si[i];
    sr[i] = r * pr - im * pi;
    si[i] = r * pi + im * pr;
  }
}

__device__ __forceinline__ void g_cz(float* sr, float* si, int i, int a, int b) {
  if ((i & wmask(a)) && (i & wmask(b))) { sr[i] = -sr[i]; si[i] = -si[i]; }
}

__device__ __forceinline__ void g_cnot(float* sr, float* si, int i, int c, int t) {
  const int mc = wmask(c), mt = wmask(t);
  __syncthreads();
  float vr = 0.f, vi = 0.f;
  if (i & mc) { int p = i ^ mt; vr = sr[p]; vi = si[p]; }
  __syncthreads();
  if (i & mc) { sr[i] = vr; si[i] = vi; }
}

__device__ __forceinline__ void g_swap(float* sr, float* si, int i, int a, int b) {
  const int ma = wmask(a), mb = wmask(b);
  __syncthreads();
  int p = (i & ~(ma | mb)) | ((i & ma) ? mb : 0) | ((i & mb) ? ma : 0);
  float vr = sr[p], vi = si[p];
  __syncthreads();
  sr[i] = vr; si[i] = vi;
}

__device__ __forceinline__ void g_toffx(float* sr, float* si, int i,
                                        int c1, int c2, int t) {
  g_h(sr, si, i, c1); g_h(sr, si, i, c2);
  g_h(sr, si, i, t);
  g_cnot(sr, si, i, c2, t); g_ph(sr, si, i, t, SQ2F, -SQ2F);
  g_cnot(sr, si, i, c1, t); g_ph(sr, si, i, t, SQ2F,  SQ2F);
  g_cnot(sr, si, i, c2, t); g_ph(sr, si, i, t, SQ2F, -SQ2F);
  g_cnot(sr, si, i, c1, t);
  g_ph(sr, si, i, c2, SQ2F, SQ2F); g_ph(sr, si, i, t, SQ2F, SQ2F);
  g_cnot(sr, si, i, c1, c2);
  g_h(sr, si, i, t);
  g_ph(sr, si, i, c1, SQ2F, SQ2F); g_ph(sr, si, i, c2, SQ2F, -SQ2F);
  g_cnot(sr, si, i, c1, c2);
  g_h(sr, si, i, c1); g_h(sr, si, i, c2);
}

// ---------------- kernel 1: fold the fixed circuit into U (512x512, f16) ----
__global__ __launch_bounds__(QDIM) void qcnn_build_u(_Float16* __restrict__ Ur,
                                                     _Float16* __restrict__ Ui) {
  __shared__ float sr[QDIM];
  __shared__ float si[QDIM];
  const int i = threadIdx.x;   // amplitude index
  const int j = blockIdx.x;    // basis column
  sr[i] = (i == j) ? 1.f : 0.f;
  si[i] = 0.f;
  __syncthreads();

  // convolution layer
  g_cz(sr, si, i, 0, 1); g_cz(sr, si, i, 2, 3);
  g_cz(sr, si, i, 4, 5); g_cz(sr, si, i, 6, 7);
  g_cz(sr, si, i, 1, 2); g_cz(sr, si, i, 3, 4);
  g_cz(sr, si, i, 5, 6); g_cz(sr, si, i, 7, 8);
  g_cz(sr, si, i, 1, 4); g_cz(sr, si, i, 4, 7);
  g_toffx(sr, si, i, 0, 2, 1);
  g_toffx(sr, si, i, 3, 5, 4);
  g_toffx(sr, si, i, 6, 8, 7);
  g_swap(sr, si, i, 2, 3); g_swap(sr, si, i, 5, 6);
  // pooling layer
  g_h(sr, si, i, 0); g_cz(sr, si, i, 0, 1);
  g_h(sr, si, i, 3); g_cz(sr, si, i, 3, 4);
  g_h(sr, si, i, 6); g_cz(sr, si, i, 6, 7);
  g_h(sr, si, i, 2); g_cz(sr, si, i, 2, 1);
  g_h(sr, si, i, 5); g_cz(sr, si, i, 5, 4);
  g_h(sr, si, i, 8); g_cz(sr, si, i, 8, 7);
  // fc layer
  g_cz(sr, si, i, 1, 4); g_cz(sr, si, i, 4, 7);

  // column j of U; store row-major so B-fragments read contiguous K
  Ur[(long)i * QDIM + j] = (_Float16)sr[i];
  Ui[(long)i * QDIM + j] = (_Float16)si[i];
}

// ---------------- WMMA fragment assembly -------------------------------------
__device__ __forceinline__ v16h mk16(v8h lo, v8h hi) {
  return __builtin_shufflevector(lo, hi, 0, 1, 2, 3, 4, 5, 6, 7,
                                         8, 9, 10, 11, 12, 13, 14, 15);
}

#define WMMA_F16(A, B, C) \
  __builtin_amdgcn_wmma_f32_16x16x32_f16(false, (A), false, (B), (short)0, (C), false, false)

// ---------------- kernel 2: batched complex GEMM + <X_4> epilogue ------------
// 32 states per block (two 16-row M-tiles share every B fragment -> 2x L2 reuse)
__global__ __launch_bounds__(128) void qcnn_expval_wmma(
    const float* __restrict__ Sr, const float* __restrict__ Si,
    const _Float16* __restrict__ Ur, const _Float16* __restrict__ Ui,
    float* __restrict__ out) {
  __shared__ _Float16 As_r[32 * QDIM];   // Re(psi) f16  (32 KB)
  __shared__ _Float16 As_i[32 * QDIM];   // Im(psi) f16  (32 KB)
  __shared__ float acc[32];

  const int tid  = threadIdx.x;
  const int lane = tid & 31;
  const int wave = tid >> 5;
  const long b0  = (long)blockIdx.x * 32;

  if (tid < 32) acc[tid] = 0.f;

  // stage A: 32 states x 512 f32 (re+im) -> f16 LDS, vectorized
  for (int v = tid; v < 32 * QDIM / 4; v += 128) {
    const int m = v >> 7;
    const int q = v & 127;
    const float4 fr = ((const float4*)(Sr + (b0 + m) * QDIM))[q];
    const float4 fi = ((const float4*)(Si + (b0 + m) * QDIM))[q];
    const int idx = m * QDIM + q * 4;
    v4h hr = {(_Float16)fr.x, (_Float16)fr.y, (_Float16)fr.z, (_Float16)fr.w};
    v4h hi = {(_Float16)fi.x, (_Float16)fi.y, (_Float16)fi.z, (_Float16)fi.w};
    *(v4h*)&As_r[idx] = hr;
    *(v4h*)&As_i[idx] = hi;
  }
  __syncthreads();

  const int m16 = lane & 15;
  float t0[8], t1[8];
#pragma unroll
  for (int r = 0; r < 8; ++r) { t0[r] = 0.f; t1[r] = 0.f; }

  // 16 column-chunks of 32 (lo: bit4=0, hi: bit4=1); wave w does chunks w,w+4,w+8,w+12
  for (int cc = 0; cc < 4; ++cc) {
    const int c   = wave + 4 * cc;
    const int nlo = c * 32 + m16;
    const int nhi = nlo + 16;
    const _Float16* Brl_p = Ur + (long)nlo * QDIM;
    const _Float16* Bil_p = Ui + (long)nlo * QDIM;
    const _Float16* Brh_p = Ur + (long)nhi * QDIM;
    const _Float16* Bih_p = Ui + (long)nhi * QDIM;

    // warm L2/WGP$ for the next chunk's B rows (global_prefetch_b8)
    if (cc < 3) {
      const long nn = (long)((wave + 4 * (cc + 1)) * 32 + m16) * QDIM;
      __builtin_prefetch(Ur + nn, 0, 1);
      __builtin_prefetch(Ui + nn, 0, 1);
      __builtin_prefetch(Ur + nn + 16 * QDIM, 0, 1);
      __builtin_prefetch(Ui + nn + 16 * QDIM, 0, 1);
    }

    v8f Crl0 = {}, Cil0 = {}, Crh0 = {}, Cih0 = {};
    v8f Crl1 = {}, Cil1 = {}, Crh1 = {}, Cih1 = {};
#pragma unroll 2
    for (int k0 = 0; k0 < QDIM; k0 += 32) {
      const int ka = (lane < 16) ? k0 : k0 + 8;   // ISA 16-bit A 16x32 layout
      // shared B fragments (L2-resident U)
      v16h Brl = mk16(*(const v8h*)(Brl_p + ka), *(const v8h*)(Brl_p + ka + 16));
      v16h Bil = mk16(*(const v8h*)(Bil_p + ka), *(const v8h*)(Bil_p + ka + 16));
      v16h Brh = mk16(*(const v8h*)(Brh_p + ka), *(const v8h*)(Brh_p + ka + 16));
      v16h Bih = mk16(*(const v8h*)(Bih_p + ka), *(const v8h*)(Bih_p + ka + 16));

      // ---- M-tile 0: states b0 .. b0+15 ----
      {
        const int ra = m16 * QDIM + ka;
        v16h Ar = mk16(*(const v8h*)&As_r[ra], *(const v8h*)&As_r[ra + 16]);
        v16h Ai = mk16(*(const v8h*)&As_i[ra], *(const v8h*)&As_i[ra + 16]);
        v16h An = -Ai;   // folds the complex subtract; co-executes with WMMA
        Crl0 = WMMA_F16(Ar, Brl, Crl0);  Crl0 = WMMA_F16(An, Bil, Crl0);
        Cil0 = WMMA_F16(Ar, Bil, Cil0);  Cil0 = WMMA_F16(Ai, Brl, Cil0);
        Crh0 = WMMA_F16(Ar, Brh, Crh0);  Crh0 = WMMA_F16(An, Bih, Crh0);
        Cih0 = WMMA_F16(Ar, Bih, Cih0);  Cih0 = WMMA_F16(Ai, Brh, Cih0);
      }
      // ---- M-tile 1: states b0+16 .. b0+31 ----
      {
        const int ra = (m16 + 16) * QDIM + ka;
        v16h Ar = mk16(*(const v8h*)&As_r[ra], *(const v8h*)&As_r[ra + 16]);
        v16h Ai = mk16(*(const v8h*)&As_i[ra], *(const v8h*)&As_i[ra + 16]);
        v16h An = -Ai;
        Crl1 = WMMA_F16(Ar, Brl, Crl1);  Crl1 = WMMA_F16(An, Bil, Crl1);
        Cil1 = WMMA_F16(Ar, Bil, Cil1);  Cil1 = WMMA_F16(Ai, Brl, Cil1);
        Crh1 = WMMA_F16(Ar, Brh, Crh1);  Crh1 = WMMA_F16(An, Bih, Crh1);
        Cih1 = WMMA_F16(Ar, Bih, Cih1);  Cih1 = WMMA_F16(Ai, Brh, Cih1);
      }
    }
    // <X_4> pairing: same (M,N) slot of lo/hi tiles are phi_i and phi_{i|16}
#pragma unroll
    for (int r = 0; r < 8; ++r) {
      t0[r] += Crl0[r] * Crh0[r] + Cil0[r] * Cih0[r];
      t1[r] += Crl1[r] * Crh1[r] + Cil1[r] * Cih1[r];
    }
  }

  // reduce over N within each 16-lane half (C layout: half0 -> M=r, half1 -> M=r+8)
#pragma unroll
  for (int r = 0; r < 8; ++r) {
    float a = t0[r], b = t1[r];
    a += __shfl_xor(a, 1, 32);  b += __shfl_xor(b, 1, 32);
    a += __shfl_xor(a, 2, 32);  b += __shfl_xor(b, 2, 32);
    a += __shfl_xor(a, 4, 32);  b += __shfl_xor(b, 4, 32);
    a += __shfl_xor(a, 8, 32);  b += __shfl_xor(b, 8, 32);
    t0[r] = a; t1[r] = b;
  }
  if (lane == 0) {
#pragma unroll
    for (int r = 0; r < 8; ++r) {
      atomicAdd(&acc[r],      t0[r]);
      atomicAdd(&acc[16 + r], t1[r]);
    }
  } else if (lane == 16) {
#pragma unroll
    for (int r = 0; r < 8; ++r) {
      atomicAdd(&acc[8 + r],  t0[r]);
      atomicAdd(&acc[24 + r], t1[r]);
    }
  }
  __syncthreads();
  if (tid < 32) out[b0 + tid] = 2.f * acc[tid];
}

// ---------------- host launch -------------------------------------------------
extern "C" void kernel_launch(void* const* d_in, const int* in_sizes, int n_in,
                              void* d_out, int out_size, void* d_ws, size_t ws_size,
                              hipStream_t stream) {
  const float* Sr = (const float*)d_in[0];
  const float* Si = (const float*)d_in[1];
  float* out = (float*)d_out;
  const int bsz = in_sizes[0] / QDIM;   // 65536

  // workspace: U real then U imag, each 512*512 f16 (1 MB total, L2-resident)
  _Float16* Ur = (_Float16*)d_ws;
  _Float16* Ui = (_Float16*)((char*)d_ws + (size_t)QDIM * QDIM * sizeof(_Float16));

  qcnn_build_u<<<QDIM, QDIM, 0, stream>>>(Ur, Ui);
  qcnn_expval_wmma<<<bsz / 32, 128, 0, stream>>>(Sr, Si, Ur, Ui, out);
}
